// FlashAttention_89197880803699
// MI455X (gfx1250) — matrix operand order repaired and checked by reading, not compiled, not run on hardware
//
#include <hip/hip_runtime.h>
#include <stdint.h>

typedef __attribute__((ext_vector_type(16))) _Float16 v16h;
typedef __attribute__((ext_vector_type(8)))  float    v8f;

#if __has_builtin(__builtin_amdgcn_exp2f)
#define FAST_EXP2(x) __builtin_amdgcn_exp2f(x)
#else
#define FAST_EXP2(x) exp2f(x)
#endif

namespace {
constexpr int   kBatch   = 2;
constexpr int   kHeads   = 16;
constexpr int   kSeq     = 2048;
constexpr int   kDim     = 64;
constexpr float kSmScale = 0.125f;
constexpr float kLog2e   = 1.44269504088896340736f;

constexpr int BLOCK_M  = 64;   // 4 waves x 16 query rows each
constexpr int BLOCK_N  = 32;   // K/V tile rows per iteration
constexpr int NTHREADS = 128;  // 4 wave32
constexpr int kBH      = kBatch * kHeads;

constexpr int SK_STRIDE = kDim + 8;      // 72 halves = 144B (16B aligned)
constexpr int SV_STRIDE = BLOCK_N + 8;   // 40 halves = 80B  (16B aligned)
}

// ---- CDNA5 async global->LDS staging (inline asm; tracked by ASYNCcnt) ----
__device__ __forceinline__ void async_b128(uint32_t l, uint64_t g) {
    asm volatile("global_load_async_to_lds_b128 %0, %1, off"
                 :: "v"(l), "v"(g) : "memory");
}
__device__ __forceinline__ void wait_async0() {
    asm volatile("s_wait_asynccnt 0" ::: "memory");
}

// ---- lane <-> lane^16 half-swap in one VALU op (v_permlanex16_b32) ----
__device__ __forceinline__ int xor16_i(int x) {
#if __has_builtin(__builtin_amdgcn_permlanex16)
    return __builtin_amdgcn_permlanex16(x, x, 0x76543210, (int)0xFEDCBA98,
                                        false, false);
#else
    return __shfl_xor(x, 16, 32);
#endif
}
__device__ __forceinline__ float xor16_f(float x) {
    return __builtin_bit_cast(float, xor16_i(__builtin_bit_cast(int, x)));
}

// =====================================================================
// Pre-pass: K (f32 row-major) -> f16 row-major; V (f32 row-major) ->
// f16 TRANSPOSED [d][n].  One 64-row tile per block, LDS-bounced transpose.
// =====================================================================
__global__ __launch_bounds__(256)
void fa_cvt_kernel(const float* __restrict__ kg, const float* __restrict__ vg,
                   _Float16* __restrict__ kh, _Float16* __restrict__ vth)
{
    __shared__ _Float16 sT[64][72];          // V tile staging [n][d]
    const int n0  = blockIdx.x * 64;
    const int bh  = blockIdx.y;
    const int tid = threadIdx.x;
    const size_t gbase = (size_t)bh * kSeq * kDim;

    {   // K: straight f32 -> f16 copy of 64x64 tile
        const float* src = kg + gbase + (size_t)n0 * kDim;
        _Float16*    dst = kh + gbase + (size_t)n0 * kDim;
        const int t0 = tid * 16;
#pragma unroll
        for (int i = 0; i < 16; i += 4) {
            const float4 f = *(const float4*)(src + t0 + i);
            _Float16* d = dst + t0 + i;
            d[0] = (_Float16)f.x; d[1] = (_Float16)f.y;
            d[2] = (_Float16)f.z; d[3] = (_Float16)f.w;
        }
    }
    {   // V: load 64x64 tile, transpose via LDS, store rows of V^T
        const float* src = vg + gbase + (size_t)n0 * kDim;
        const int t0 = tid * 16;
#pragma unroll
        for (int i = 0; i < 16; i += 4) {
            const int idx = t0 + i;
            const int n = idx >> 6, d = idx & 63;
            const float4 f = *(const float4*)(src + idx);
            sT[n][d + 0] = (_Float16)f.x; sT[n][d + 1] = (_Float16)f.y;
            sT[n][d + 2] = (_Float16)f.z; sT[n][d + 3] = (_Float16)f.w;
        }
        __syncthreads();
        const int d = tid >> 2, part = tid & 3;   // 64 rows x 4 chunks of 16
        _Float16* dst = vth + (size_t)bh * kDim * kSeq + (size_t)d * kSeq
                            + n0 + part * 16;
#pragma unroll
        for (int i = 0; i < 16; ++i) dst[i] = sT[part * 16 + i][d];
    }
}

// =====================================================================
// Transposed-domain flash attention, double-buffered async pipeline:
//   S^T = K * Q^T ; per-query softmax state lives per-LANE;
//   rowsum accumulated by ones-WMMA ; O^T += V^T * P^T.
// =====================================================================
__global__ __launch_bounds__(NTHREADS)
void fa_fwd_wmma_kernel(const float* __restrict__ qg,
                        const _Float16* __restrict__ kh,
                        const _Float16* __restrict__ vth,
                        float* __restrict__ og)
{
    __shared__ _Float16 sK [2][BLOCK_N][SK_STRIDE];   // [buf][n][d]
    __shared__ _Float16 sVt[2][kDim][SV_STRIDE];      // [buf][d][n]

    const int tid  = threadIdx.x;
    const int wave = tid >> 5;
    const int lane = tid & 31;
    const int l16  = lane & 15;
    const bool hi  = (lane >= 16);
    const int kb8  = hi ? 8 : 0;    // unused; A-fragments also use kb16
    const int kb16 = hi ? 16 : 0;   // B-fragment K-half base

    const int qblk = blockIdx.x;
    const int bh   = blockIdx.y;
    const size_t base = (size_t)bh * kSeq * kDim;
    const _Float16* kbh = kh  + base;
    const _Float16* vbh = vth + base;   // [d][kSeq]

    const int q0     = qblk * BLOCK_M + wave * 16;
    const int m_glob = q0 + l16;        // this lane's query row

    // ---- per-thread DMA chunk coords (2 chunks each for K and V^T) ----
    const int c0  = tid * 2;
    const int kr0 = (c0)     >> 3, kc0 = (c0)     & 7;
    const int kr1 = (c0 + 1) >> 3, kc1 = (c0 + 1) & 7;
    const int vd0 = (c0)     >> 2, vc0 = (c0)     & 3;
    const int vd1 = (c0 + 1) >> 2, vc1 = (c0 + 1) & 3;

    // LDS chunk byte addresses per buffer (computed once)
    uint32_t lK0[2], lK1[2], lV0[2], lV1[2];
#pragma unroll
    for (int b = 0; b < 2; ++b) {
        lK0[b] = (uint32_t)(uintptr_t)&sK [b][kr0][kc0 * 8];
        lK1[b] = (uint32_t)(uintptr_t)&sK [b][kr1][kc1 * 8];
        lV0[b] = (uint32_t)(uintptr_t)&sVt[b][vd0][vc0 * 8];
        lV1[b] = (uint32_t)(uintptr_t)&sVt[b][vd1][vc1 * 8];
    }
    // global chunk addresses, incremented by constant tile steps
    uint64_t gK0 = (uint64_t)(uintptr_t)(kbh + kr0 * kDim + kc0 * 8);
    uint64_t gK1 = (uint64_t)(uintptr_t)(kbh + kr1 * kDim + kc1 * 8);
    uint64_t gV0 = (uint64_t)(uintptr_t)(vbh + (size_t)vd0 * kSeq + vc0 * 8);
    uint64_t gV1 = (uint64_t)(uintptr_t)(vbh + (size_t)vd1 * kSeq + vc1 * 8);
    constexpr uint64_t KSTEP = (uint64_t)BLOCK_N * kDim * sizeof(_Float16); // 4096 B
    constexpr uint64_t VSTEP = (uint64_t)BLOCK_N * sizeof(_Float16);        //   64 B

    // ---- Q^T B-fragments (pre-scaled into log2 domain), loaded once ----
    const float qscale = kSmScale * kLog2e;
    const float* qrow  = qg + base + (size_t)m_glob * kDim;
    v16h bQ0, bQ1;
#pragma unroll
    for (int i = 0; i < 16; ++i) {
        bQ0[i] = (_Float16)(qrow[kb16 + i]      * qscale);
        bQ1[i] = (_Float16)(qrow[32 + kb16 + i] * qscale);
    }

    v16h aOnes;
#pragma unroll
    for (int i = 0; i < 16; ++i) aOnes[i] = (_Float16)1.0f;

    v8f acc[4];
#pragma unroll
    for (int t = 0; t < 4; ++t) acc[t] = v8f{};
    v8f   lsum = v8f{};     // rowsum accumulator (all elements equal)
    float m_s  = -1e30f;

    const int q_hi_row = qblk * BLOCK_M + BLOCK_M - 1;
    const int nblocks  = q_hi_row / BLOCK_N + 1;

    // issue DMA for the next tile; addresses advance by constant steps
    auto stage = [&](int buf) {
        async_b128(lK0[buf], gK0);  gK0 += KSTEP;
        async_b128(lK1[buf], gK1);  gK1 += KSTEP;
        async_b128(lV0[buf], gV0);  gV0 += VSTEP;
        async_b128(lV1[buf], gV1);  gV1 += VSTEP;
    };

    stage(0);   // prologue: tile 0

    for (int j = 0; j < nblocks; ++j) {
        const int n0  = j * BLOCK_N;
        const int buf = j & 1;

        wait_async0();        // my chunks of tile j are in LDS
        __syncthreads();      // everyone's chunks are in LDS; prev reads done

        if (j + 1 < nblocks) stage(buf ^ 1);   // overlap next tile's DMA

        // ---------- S^T = K * Q^T : two 16x16 tiles ----------
        v8f sc[2];
#pragma unroll
        for (int t = 0; t < 2; ++t) {
            const _Float16* kr = &sK[buf][t * 16 + l16][0];
            v16h a0, a1;
#pragma unroll
            for (int i = 0; i < 8; ++i) {
                a0[i]     = kr[kb16 + i];
                a0[8 + i] = kr[kb16 + 8 + i];
                a1[i]     = kr[32 + kb16 + i];
                a1[8 + i] = kr[32 + kb16 + 8 + i];
            }
            v8f c = v8f{};
            c = __builtin_amdgcn_wmma_f32_16x16x32_f16(false, a0, false, bQ0,
                                                       (short)0, c, false, false);
            c = __builtin_amdgcn_wmma_f32_16x16x32_f16(false, a1, false, bQ1,
                                                       (short)0, c, false, false);
            sc[t] = c;
        }

        // ---------- causal mask ----------
        if (n0 + BLOCK_N - 1 > q0) {
            const int nb = n0 + (hi ? 8 : 0);
#pragma unroll
            for (int t = 0; t < 2; ++t)
#pragma unroll
                for (int r = 0; r < 8; ++r)
                    sc[t][r] = (nb + t * 16 + r <= m_glob) ? sc[t][r] : -1e30f;
        }

        // ---------- online softmax: per-lane stats ----------
        float mx = sc[0][0];
#pragma unroll
        for (int r = 1; r < 8; ++r) mx = fmaxf(mx, sc[0][r]);
#pragma unroll
        for (int r = 0; r < 8; ++r) mx = fmaxf(mx, sc[1][r]);
        mx = fmaxf(mx, xor16_f(mx));

        if (__any(mx > m_s)) {          // rescale only when the max moved
            const float mnew  = fmaxf(m_s, mx);
            const float alpha = FAST_EXP2(m_s - mnew);
            m_s = mnew;
#pragma unroll
            for (int r = 0; r < 8; ++r) lsum[r] *= alpha;
#pragma unroll
            for (int td = 0; td < 4; ++td)
#pragma unroll
                for (int r = 0; r < 8; ++r) acc[td][r] *= alpha;
        }
#pragma unroll
        for (int r = 0; r < 8; ++r) {
            sc[0][r] = FAST_EXP2(sc[0][r] - m_s);
            sc[1][r] = FAST_EXP2(sc[1][r] - m_s);
        }

        // ---------- P^T C-layout -> B-fragment via fixed half-swap ----------
        union H8  { _Float16 h[8];  int u[4]; } p0, p1;
        union B16 { v16h v;         int u[8]; } bP;
#pragma unroll
        for (int r = 0; r < 8; ++r) { p0.h[r] = (_Float16)sc[0][r]; p1.h[r] = (_Float16)sc[1][r]; }
#pragma unroll
        for (int i = 0; i < 4; ++i) {
            const int o0 = xor16_i(p0.u[i]);
            const int o1 = xor16_i(p1.u[i]);
            bP.u[i]     = hi ? o1      : p0.u[i];
            bP.u[4 + i] = hi ? p1.u[i] : o0;
        }

        // ---------- rowsum accumulated by ones-WMMA (no VALU consumer) ----------
        lsum = __builtin_amdgcn_wmma_f32_16x16x32_f16(false, aOnes, false, bP.v,
                                                      (short)0, lsum, false, false);

        // ---------- O^T += V^T * P^T  (4 d-tiles) ----------
#pragma unroll
        for (int td = 0; td < 4; ++td) {
            const _Float16* vr = &sVt[buf][td * 16 + l16][0];
            v16h aV;
#pragma unroll
            for (int i = 0; i < 8; ++i) { aV[i] = vr[kb16 + i]; aV[8 + i] = vr[kb16 + 8 + i]; }
            acc[td] = __builtin_amdgcn_wmma_f32_16x16x32_f16(false, aV, false, bP.v,
                                                             (short)0, acc[td], false, false);
        }
    }

    // ---------- epilogue ----------
    const float inv = 1.0f / lsum[0];
    float* op = og + base + (size_t)m_glob * kDim + (hi ? 8 : 0);
#pragma unroll
    for (int td = 0; td < 4; ++td) {
        float4 lo  = make_float4(acc[td][0] * inv, acc[td][1] * inv,
                                 acc[td][2] * inv, acc[td][3] * inv);
        float4 hi4 = make_float4(acc[td][4] * inv, acc[td][5] * inv,
                                 acc[td][6] * inv, acc[td][7] * inv);
        *(float4*)(op + td * 16)     = lo;
        *(float4*)(op + td * 16 + 4) = hi4;
    }
}

extern "C" void kernel_launch(void* const* d_in, const int* in_sizes, int n_in,
                              void* d_out, int out_size, void* d_ws, size_t ws_size,
                              hipStream_t stream) {
    const float* q = (const float*)d_in[0];
    const float* k = (const float*)d_in[1];
    const float* v = (const float*)d_in[2];
    float* out = (float*)d_out;
    (void)in_sizes; (void)n_in; (void)out_size; (void)ws_size;

    _Float16* kh  = (_Float16*)d_ws;                         // [bh][n][d] f16
    _Float16* vth = kh + (size_t)kBH * kSeq * kDim;          // [bh][d][n] f16

    fa_cvt_kernel<<<dim3(kSeq / 64, kBH), 256, 0, stream>>>(k, v, kh, vth);
    fa_fwd_wmma_kernel<<<dim3(kSeq / BLOCK_M, kBH), NTHREADS, 0, stream>>>(q, kh, vth, out);
}
